// RBFLayer_2430951489991
// MI455X (gfx1250) — matrix operand order repaired and checked by reading, not compiled, not run on hardware
//
#include <hip/hip_runtime.h>

// RBF layer: out[b,k] = exp(-beta[k] * ||x[b]-c[k]||^2), B=65536, D=64, K=512.
// Split-bf16 WMMA GEMM for the cross term (-2 x.cT) with exact f32 x^2/c^2
// correction; fused exp epilogue. HBM-write bound (~151 MB -> ~6.5us floor).
// Two n-tiles are interleaved per iteration (independent accumulators) to fill
// the bf16 WMMA hazard slots and overlap LDS loads with matrix ops.

typedef __attribute__((ext_vector_type(16))) __bf16 v16bf;
typedef __attribute__((ext_vector_type(8)))  __bf16 v8bf;
typedef __attribute__((ext_vector_type(8)))  float  v8f;

#define B_TOT 65536
#define D_DIM 64
#define K_CTR 512
#define DP    72   // padded LDS row stride in halves (144B: 16B-aligned, fewer bank conflicts)

static __device__ __forceinline__ v16bf lds_load16(const __bf16* p) {
  v8bf a = *(const v8bf*)(p);
  v8bf b = *(const v8bf*)(p + 16);
  v16bf r;
  #pragma unroll
  for (int j = 0; j < 8; ++j) { r[j] = a[j]; r[j + 8] = b[j]; }
  return r;
}

__launch_bounds__(256)
__global__ void rbf_wmma_kernel(const float* __restrict__ x,
                                const float* __restrict__ centers,
                                const float* __restrict__ betas,
                                float* __restrict__ out) {
  extern __shared__ char smem[];
  __bf16* chi = (__bf16*)smem;                 // K_CTR * DP bf16 (hi part)
  __bf16* clo = chi + K_CTR * DP;              // K_CTR * DP bf16 (lo part)
  float*  c2s = (float*)(clo + K_CTR * DP);    // K_CTR f32  (||c||^2)
  float*  bts = c2s + K_CTR;                   // K_CTR f32  (betas)

  const int tid = threadIdx.x;

  // ---- Phase 1: convert centers -> bf16 hi/lo into LDS (B-operand layout),
  //      compute c^2, stage betas. Centers are only 131 KB -> L2 resident.
  #pragma unroll
  for (int rr = 0; rr < 2; ++rr) {
    const int k = tid * 2 + rr;                // 256 threads x 2 rows = 512
    const float* crow = centers + k * D_DIM;
    float s = 0.0f;
    #pragma unroll
    for (int d = 0; d < D_DIM; d += 2) {
      float2 v = *(const float2*)(crow + d);
      s += v.x * v.x + v.y * v.y;
      __bf16 h0 = (__bf16)v.x;
      __bf16 h1 = (__bf16)v.y;
      chi[k * DP + d]     = h0;
      chi[k * DP + d + 1] = h1;
      clo[k * DP + d]     = (__bf16)(v.x - (float)h0);
      clo[k * DP + d + 1] = (__bf16)(v.y - (float)h1);
    }
    c2s[k] = s;
  }
  bts[tid]       = betas[tid];
  bts[tid + 256] = betas[tid + 256];
  __syncthreads();

  // ---- Phase 2: each wave computes a 16-row x 512-col strip.
  const int lane  = tid & 31;
  const int wave  = tid >> 5;
  const int lhalf = lane >> 4;                 // 0: lanes 0-15, 1: lanes 16-31
  const int l15   = lane & 15;
  const int mrow  = blockIdx.x * 128 + wave * 16 + l15;   // row this lane feeds into A
  const float* xrow = x + mrow * D_DIM;

  // Build A operands (16x32 bf16 layout: lanes<16 hold K=off+0..7,16..23;
  // lanes>=16 hold K=off+8..15,24..31; element j -> VGPR j/2, half j%2).
  v16bf a_hi[2], a_lo[2];
  float ssq = 0.0f;
  #pragma unroll
  for (int c = 0; c < 2; ++c) {
    const int dbase = c * 32 + lhalf * 8;
    float4 r0a = *(const float4*)(xrow + dbase);
    float4 r0b = *(const float4*)(xrow + dbase + 4);
    float4 r1a = *(const float4*)(xrow + dbase + 16);
    float4 r1b = *(const float4*)(xrow + dbase + 20);
    float v[16] = {r0a.x, r0a.y, r0a.z, r0a.w, r0b.x, r0b.y, r0b.z, r0b.w,
                   r1a.x, r1a.y, r1a.z, r1a.w, r1b.x, r1b.y, r1b.z, r1b.w};
    #pragma unroll
    for (int j = 0; j < 16; ++j) {
      ssq += v[j] * v[j];
      __bf16 h = (__bf16)v[j];
      a_hi[c][j] = h;
      a_lo[c][j] = (__bf16)(v[j] - (float)h);
    }
  }
  // Lanes L and L^16 together hold all 64 elements of row l15.
  ssq += __shfl_xor(ssq, 16, 32);
  // C-tile: lane element r corresponds to M = r + lhalf*8 -> fetch its x^2.
  float x2v[8];
  #pragma unroll
  for (int r = 0; r < 8; ++r)
    x2v[r] = __shfl(ssq, lhalf * 8 + r, 32);

  const int rowStoreBase = blockIdx.x * 128 + wave * 16 + lhalf * 8;

  #pragma unroll 1
  for (int t = 0; t < 32; t += 2) {
    const int n0 = t * 16 + l15;               // center (column) index, tile 0
    const int n1 = n0 + 16;                    // tile 1
    // Hoist scalar-per-column LDS reads so they overlap the WMMA chains.
    const float c2n0 = c2s[n0], bt0 = bts[n0];
    const float c2n1 = c2s[n1], bt1 = bts[n1];

    v8f acc0 = {}, acc1 = {};
    #pragma unroll
    for (int c = 0; c < 2; ++c) {
      // B operand (32x16 bf16): lanes<16 hold K=c*32+0..7,16..23; lanes>=16 hold +8..15,24..31.
      const int doff = c * 32 + (lhalf ? 8 : 0);
      v16bf bh0 = lds_load16(chi + n0 * DP + doff);
      v16bf bl0 = lds_load16(clo + n0 * DP + doff);
      v16bf bh1 = lds_load16(chi + n1 * DP + doff);
      v16bf bl1 = lds_load16(clo + n1 * DP + doff);
      // cross += x_hi*c_hi + x_hi*c_lo + x_lo*c_hi  (f32 accumulate),
      // alternated across the two independent accumulators to fill hazard slots.
      acc0 = __builtin_amdgcn_wmma_f32_16x16x32_bf16(false, a_hi[c], false, bh0,
                                                     (short)0, acc0, false, false);
      acc1 = __builtin_amdgcn_wmma_f32_16x16x32_bf16(false, a_hi[c], false, bh1,
                                                     (short)0, acc1, false, false);
      acc0 = __builtin_amdgcn_wmma_f32_16x16x32_bf16(false, a_hi[c], false, bl0,
                                                     (short)0, acc0, false, false);
      acc1 = __builtin_amdgcn_wmma_f32_16x16x32_bf16(false, a_hi[c], false, bl1,
                                                     (short)0, acc1, false, false);
      acc0 = __builtin_amdgcn_wmma_f32_16x16x32_bf16(false, a_lo[c], false, bh0,
                                                     (short)0, acc0, false, false);
      acc1 = __builtin_amdgcn_wmma_f32_16x16x32_bf16(false, a_lo[c], false, bh1,
                                                     (short)0, acc1, false, false);
    }
    #pragma unroll
    for (int r = 0; r < 8; ++r) {
      float d2a = x2v[r] + c2n0 - 2.0f * acc0[r];
      float d2b = x2v[r] + c2n1 - 2.0f * acc1[r];
      d2a = fmaxf(d2a, 0.0f);
      d2b = fmaxf(d2b, 0.0f);
      const size_t rowOff = (size_t)(rowStoreBase + r) * K_CTR;
      out[rowOff + n0] = __expf(-bt0 * d2a);
      out[rowOff + n1] = __expf(-bt1 * d2b);
    }
  }
}

extern "C" void kernel_launch(void* const* d_in, const int* in_sizes, int n_in,
                              void* d_out, int out_size, void* d_ws, size_t ws_size,
                              hipStream_t stream) {
  (void)in_sizes; (void)n_in; (void)out_size; (void)d_ws; (void)ws_size;
  const float* x       = (const float*)d_in[0];
  const float* centers = (const float*)d_in[1];
  const float* betas   = (const float*)d_in[2];
  float* out = (float*)d_out;

  const size_t smem = (size_t)K_CTR * DP * 2 * sizeof(__bf16)  // chi + clo
                    + (size_t)K_CTR * 2 * sizeof(float);       // c2 + betas
  dim3 grid(B_TOT / 128);   // 512 workgroups, each owns 128 rows
  dim3 block(256);          // 8 wave32 waves
  rbf_wmma_kernel<<<grid, block, smem, stream>>>(x, centers, betas, out);
}
